// ResnetB_63969242906671
// MI455X (gfx1250) — matrix operand order repaired and checked by reading, not compiled, not run on hardware
//
#include <hip/hip_runtime.h>
#include <math.h>

#define N_PTS 16384
#define KNB   32
#define NKP   15
#define CIN   256
#define CMID  128
#define COUT  512
#define K2    2048              // padded (16 kernel pts) * CMID
#define EXTENTF 0.8f
#define BN_EPSF 1e-5f
#define SLOPEF  0.1f

typedef unsigned short u16;
typedef unsigned int   u32;
typedef __attribute__((ext_vector_type(16))) __bf16 v16bf;
typedef __attribute__((ext_vector_type(8)))  float  v8f;

union BF16x16 { u16 u[16]; v16bf v; };

__device__ __forceinline__ u16 f2bf(float f) {
  u32 u = __float_as_uint(f);
  u32 r = u + 0x7FFFu + ((u >> 16) & 1u);   // round-to-nearest-even
  return (u16)(r >> 16);
}

// A-operand local-K index for VGPR pair i (0..7); hi = (lane>=16)
__device__ __forceinline__ int kA(int i, int hi) {
  return (i < 4 ? 2 * i : 16 + 2 * (i - 4)) + (hi ? 8 : 0);
}
// B-operand local-K index for VGPR pair i
__device__ __forceinline__ int kB(int i, int hi) {
  return (i < 4 ? 2 * i : 16 + 2 * (i - 4)) + (hi ? 8 : 0);
}

// gfx1250 async copy: global -> LDS, 16B per lane, tracked with ASYNCcnt.
__device__ __forceinline__ void async_g2l_b128(u32 lds_off, const void* gaddr) {
  asm volatile("global_load_async_to_lds_b128 %0, %1, off"
               :: "v"(lds_off), "v"(gaddr) : "memory");
}
__device__ __forceinline__ void wait_asynccnt0() {
  asm volatile("s_wait_asynccnt 0" ::: "memory");
}

// ---------------------------------------------------------------------------
// Pack weights to bf16 in B-operand-tiled layout [(kk*NC + col)*32 + klocal],
// pad kp_weights P 15->16 with zeros, zero the BN stats accumulators.
// ---------------------------------------------------------------------------
__global__ void k_prep(const float* __restrict__ Wi, const float* __restrict__ Wo,
                       const float* __restrict__ Kw,
                       u16* __restrict__ WiP, u16* __restrict__ WoP,
                       u16* __restrict__ KpP, float* __restrict__ stats) {
  int o = blockIdx.x * blockDim.x + threadIdx.x;
  if (o < 32768) {                       // W_in: K=256, N=128
    int kl = o & 31; int q = o >> 5; int col = q & 127; int kk = q >> 7;
    WiP[o] = f2bf(Wi[(kk * 32 + kl) * CMID + col]);
    return;
  }
  int o2 = o - 32768;
  if (o2 < 65536) {                      // W_out: K=128, N=512
    int kl = o2 & 31; int q = o2 >> 5; int col = q & 511; int kk = q >> 9;
    WoP[o2] = f2bf(Wo[(kk * 32 + kl) * COUT + col]);
    return;
  }
  int o3 = o2 - 65536;
  if (o3 < 262144) {                     // kp_weights: K=2048 (padded), N=128
    int kl = o3 & 31; int q = o3 >> 5; int d = q & 127; int kk = q >> 7;
    int r = kk * 32 + kl;                // r = p*128 + c
    float v = 0.0f;
    if (r < NKP * CMID) {
      int p = r >> 7, c = r & 127;
      v = Kw[((size_t)p * CMID + c) * CMID + d];
    }
    KpP[o3] = f2bf(v);
    return;
  }
  int o4 = o3 - 262144;
  if (o4 < 2048) stats[o4] = 0.0f;       // 2 branches * (sum[512], sumsq[512])
}

// ---------------------------------------------------------------------------
// H[N,128](bf16) = bf16(X[N,256]) @ WiP + b_in.   One 16x16 tile per wave.
// ---------------------------------------------------------------------------
__global__ void __launch_bounds__(128)
k_gemm_in(const float* __restrict__ X, const u16* __restrict__ Wp,
          const float* __restrict__ bias, u16* __restrict__ H) {
  int lane = threadIdx.x & 31;
  int hi = lane >> 4, ln = lane & 15;
  int tile = blockIdx.x * 4 + (threadIdx.x >> 5);
  int mt = tile >> 3;                    // 1024 row tiles
  int nt = tile & 7;                     // 8 col tiles
  int row = mt * 16 + ln;
  int col = nt * 16 + ln;
  v8f acc = {};
  for (int kk = 0; kk < CIN / 32; ++kk) {
    BF16x16 a, b;
    const float* Xr = X + (size_t)row * CIN + kk * 32;
    const u16* Bp = Wp + ((size_t)kk * CMID + col) * 32;
#pragma unroll
    for (int i = 0; i < 8; ++i) {
      int ka = kA(i, hi);
      a.u[2 * i]     = f2bf(Xr[ka]);
      a.u[2 * i + 1] = f2bf(Xr[ka + 1]);
      int kb = kB(i, hi);
      b.u[2 * i]     = Bp[kb];
      b.u[2 * i + 1] = Bp[kb + 1];
    }
    acc = __builtin_amdgcn_wmma_f32_16x16x32_bf16(false, a.v, false, b.v,
                                                  (short)0, acc, false, false);
  }
  float bc = bias[col];
#pragma unroll
  for (int i = 0; i < 8; ++i) {
    int r = mt * 16 + i + hi * 8;
    H[(size_t)r * CMID + col] = f2bf(acc[i] + bc);
  }
}

// ---------------------------------------------------------------------------
// Fused KPConv: per block = 16 points, 128 threads (4 waves), ~216 KB LDS.
// Phase 0: influences [16pt][16P][32K] + async neighbor-feature gather
// Phase 1: agg = infl^T(16x32) @ nf(32x128) per point  (8 WMMA / point)
// Phase 2: out[16pt,128] = agg[16pt, 2048] @ KpP[2048,128]  (512 WMMA / block)
// ---------------------------------------------------------------------------
__global__ void __launch_bounds__(128)
k_kpconv(const u16* __restrict__ H, const float* __restrict__ coords,
         const int* __restrict__ neighb, const float* __restrict__ kpts,
         const u16* __restrict__ Kp, u16* __restrict__ G) {
  extern __shared__ char smem[];
  u16*   s_nf   = (u16*)(smem);                      // [16][32][128]  131072 B
  u16*   s_infl = (u16*)(smem + 131072);             // [16][16][32]    16384 B
  u16*   s_agg  = (u16*)(smem + 147456);             // [16][2048]      65536 B
  u16*   s_kpw  = (u16*)(smem + 212992);             // [128][32]        8192 B
  float* s_kp   = (float*)(smem + 221184);           // [15][3]           256 B

  int tid = threadIdx.x;
  int base = blockIdx.x * 16;

  if (tid < NKP * 3) s_kp[tid] = kpts[tid];
  __syncthreads();

  // -- Phase 0: influences + async gather ----------------------------------
  for (int t = tid; t < 512; t += 128) {
    int pt = t >> 5, k = t & 31;
    int n = base + pt;
    int nb = neighb[n * KNB + k];
    // neighbor feature row: 256 B global -> LDS, no VGPR round-trip
    const char* src = (const char*)(H + (size_t)nb * CMID);
    u32 dst = (u32)((uintptr_t)(s_nf + (size_t)(pt * 32 + k) * CMID) -
                    (uintptr_t)smem);
#pragma unroll
    for (int j = 0; j < 16; ++j)
      async_g2l_b128(dst + j * 16, src + j * 16);

    float cx = coords[n * 3 + 0], cy = coords[n * 3 + 1], cz = coords[n * 3 + 2];
    float rx = coords[nb * 3 + 0] - cx;
    float ry = coords[nb * 3 + 1] - cy;
    float rz = coords[nb * 3 + 2] - cz;
#pragma unroll
    for (int p = 0; p < NKP; ++p) {
      float dx = rx - s_kp[p * 3 + 0];
      float dy = ry - s_kp[p * 3 + 1];
      float dz = rz - s_kp[p * 3 + 2];
      float d = sqrtf(dx * dx + dy * dy + dz * dz);
      float w = fmaxf(0.0f, 1.0f - d * (1.0f / EXTENTF));
      s_infl[(pt * 16 + p) * 32 + k] = f2bf(w);
    }
    s_infl[(pt * 16 + 15) * 32 + k] = 0;             // pad row P=15
  }
  wait_asynccnt0();
  __syncthreads();

  // -- Phase 1: per-point agg ----------------------------------------------
  int lane = tid & 31, wave = tid >> 5;
  int hi = lane >> 4, ln = lane & 15;
  for (int t = wave; t < 128; t += 4) {              // (pt, ctile) pairs
    int pt = t >> 3, ct = t & 7;
    int col = ct * 16 + ln;
    BF16x16 a, b;
#pragma unroll
    for (int i = 0; i < 8; ++i) {
      int ka = kA(i, hi);
      u32 pr = *(const u32*)&s_infl[(pt * 16 + ln) * 32 + ka];
      a.u[2 * i] = (u16)pr; a.u[2 * i + 1] = (u16)(pr >> 16);
      int kb = kB(i, hi);
      b.u[2 * i]     = s_nf[(size_t)(pt * 32 + kb) * CMID + col];
      b.u[2 * i + 1] = s_nf[(size_t)(pt * 32 + kb + 1) * CMID + col];
    }
    v8f acc = {};
    acc = __builtin_amdgcn_wmma_f32_16x16x32_bf16(false, a.v, false, b.v,
                                                  (short)0, acc, false, false);
#pragma unroll
    for (int i = 0; i < 8; ++i) {
      int p = i + hi * 8;
      s_agg[pt * K2 + p * CMID + col] = f2bf(acc[i]);
    }
  }
  __syncthreads();

  // -- Phase 2: [16 x 2048] @ [2048 x 128] ---------------------------------
  v8f acc0 = {}, acc1 = {};
  int col0 = (wave * 2 + 0) * 16 + ln;
  int col1 = (wave * 2 + 1) * 16 + ln;
  u32 kpw_base = (u32)((uintptr_t)s_kpw - (uintptr_t)smem);
  for (int kk = 0; kk < K2 / 32; ++kk) {
    // stage 8 KB kp tile (already in B layout [col][klocal]) via async DMA
    {
      const char* src = (const char*)(Kp + (size_t)kk * 4096);
      for (int j = tid; j < 512; j += 128)
        async_g2l_b128(kpw_base + j * 16, src + (size_t)j * 16);
    }
    if (kk + 1 < K2 / 32)                            // -> global_prefetch_b8
      __builtin_prefetch((const void*)(Kp + (size_t)(kk + 1) * 4096 + (tid & 31) * 128), 0, 1);
    wait_asynccnt0();
    __syncthreads();

    BF16x16 a;
#pragma unroll
    for (int i = 0; i < 8; ++i) {                    // A rows = points (ln)
      int ka = kk * 32 + kA(i, hi);
      u32 pr = *(const u32*)&s_agg[ln * K2 + ka];
      a.u[2 * i] = (u16)pr; a.u[2 * i + 1] = (u16)(pr >> 16);
    }
    BF16x16 b0, b1;
#pragma unroll
    for (int i = 0; i < 8; ++i) {
      int kb = kB(i, hi);
      u32 p0 = *(const u32*)&s_kpw[col0 * 32 + kb];
      u32 p1 = *(const u32*)&s_kpw[col1 * 32 + kb];
      b0.u[2 * i] = (u16)p0; b0.u[2 * i + 1] = (u16)(p0 >> 16);
      b1.u[2 * i] = (u16)p1; b1.u[2 * i + 1] = (u16)(p1 >> 16);
    }
    acc0 = __builtin_amdgcn_wmma_f32_16x16x32_bf16(false, a.v, false, b0.v,
                                                   (short)0, acc0, false, false);
    acc1 = __builtin_amdgcn_wmma_f32_16x16x32_bf16(false, a.v, false, b1.v,
                                                   (short)0, acc1, false, false);
    __syncthreads();
  }
#pragma unroll
  for (int i = 0; i < 8; ++i) {
    int pt = i + hi * 8;
    G[(size_t)(base + pt) * CMID + col0] = f2bf(acc0[i]);
    G[(size_t)(base + pt) * CMID + col1] = f2bf(acc1[i]);
  }
}

// ---------------------------------------------------------------------------
// Y[N,512](f32) = G[N,128](bf16) @ WoP + b_out; accumulate BN sum/sumsq.
// ---------------------------------------------------------------------------
__global__ void __launch_bounds__(128)
k_gemm_out(const u16* __restrict__ G, const u16* __restrict__ Wp,
           const float* __restrict__ bias, float* __restrict__ Y,
           float* __restrict__ stats) {
  int lane = threadIdx.x & 31;
  int hi = lane >> 4, ln = lane & 15;
  int tile = blockIdx.x * 4 + (threadIdx.x >> 5);
  int mt = tile >> 5;                    // 1024 row tiles
  int nt = tile & 31;                    // 32 col tiles
  int row = mt * 16 + ln;
  int col = nt * 16 + ln;
  v8f acc = {};
  for (int kk = 0; kk < CMID / 32; ++kk) {
    BF16x16 a, b;
    const u16* Ar = G + (size_t)row * CMID + kk * 32;
    const u16* Bp = Wp + ((size_t)kk * COUT + col) * 32;
#pragma unroll
    for (int i = 0; i < 8; ++i) {
      int ka = kA(i, hi);
      a.u[2 * i] = Ar[ka]; a.u[2 * i + 1] = Ar[ka + 1];
      int kb = kB(i, hi);
      b.u[2 * i] = Bp[kb]; b.u[2 * i + 1] = Bp[kb + 1];
    }
    acc = __builtin_amdgcn_wmma_f32_16x16x32_bf16(false, a.v, false, b.v,
                                                  (short)0, acc, false, false);
  }
  float bc = bias[col];
  float s = 0.0f, s2 = 0.0f;
#pragma unroll
  for (int i = 0; i < 8; ++i) {
    int r = mt * 16 + i + hi * 8;
    float v = acc[i] + bc;
    Y[(size_t)r * COUT + col] = v;
    s += v; s2 += v * v;
  }
  s  += __shfl_down(s, 16);              // wave32: combine lane halves
  s2 += __shfl_down(s2, 16);
  if (!hi) {
    atomicAdd(&stats[col], s);
    atomicAdd(&stats[COUT + col], s2);
  }
}

// ---------------------------------------------------------------------------
__global__ void k_bn(float* __restrict__ Y, const float* __restrict__ stats,
                     const float* __restrict__ gamma, const float* __restrict__ beta) {
  int idx = blockIdx.x * blockDim.x + threadIdx.x;
  if (idx >= N_PTS * COUT) return;
  int c = idx & (COUT - 1);
  float mu  = stats[c] * (1.0f / N_PTS);
  float var = stats[COUT + c] * (1.0f / N_PTS) - mu * mu;
  float x = Y[idx];
  float h = (x - mu) * rsqrtf(var + BN_EPSF) * gamma[c] + beta[c];
  Y[idx] = h >= 0.0f ? h : SLOPEF * h;
}

__global__ void k_copy_coords(const float* __restrict__ sc,
                              const float* __restrict__ tc,
                              float* __restrict__ dst) {
  int i = blockIdx.x * blockDim.x + threadIdx.x;
  if (i < N_PTS * 3) dst[i] = sc[i];
  else if (i < 2 * N_PTS * 3) dst[i] = tc[i - N_PTS * 3];
}

// ---------------------------------------------------------------------------
extern "C" void kernel_launch(void* const* d_in, const int* in_sizes, int n_in,
                              void* d_out, int out_size, void* d_ws, size_t ws_size,
                              hipStream_t stream) {
  (void)in_sizes; (void)n_in; (void)out_size; (void)ws_size;
  const float* src        = (const float*)d_in[0];
  const float* tgt        = (const float*)d_in[1];
  const float* src_coords = (const float*)d_in[2];
  const float* tgt_coords = (const float*)d_in[3];
  const int*   src_nb     = (const int*)d_in[4];
  const int*   tgt_nb     = (const int*)d_in[5];
  const float* W_in       = (const float*)d_in[6];
  const float* b_in       = (const float*)d_in[7];
  const float* kpts       = (const float*)d_in[8];
  const float* kpw        = (const float*)d_in[9];
  const float* W_out      = (const float*)d_in[10];
  const float* b_out      = (const float*)d_in[11];
  const float* gamma      = (const float*)d_in[12];
  const float* beta       = (const float*)d_in[13];

  char* ws = (char*)d_ws;
  u16*   WiP   = (u16*)(ws);                         // 65536 B
  u16*   WoP   = (u16*)(ws + 65536);                 // 131072 B
  u16*   KpP   = (u16*)(ws + 196608);                // 524288 B
  float* stats = (float*)(ws + 720896);              // 8192 B
  char*  hb    = ws + 729088;
  const size_t HB = (size_t)N_PTS * CMID * sizeof(u16);   // 4 MB
  u16* Hb[2] = { (u16*)hb,            (u16*)(hb + HB) };
  u16* Gb[2] = { (u16*)(hb + 2 * HB), (u16*)(hb + 3 * HB) };

  float* Y = (float*)d_out;
  float* outs[2] = { Y, Y + (size_t)N_PTS * COUT };
  float* cdst = Y + 2 * (size_t)N_PTS * COUT;

  k_prep<<<(362496 + 255) / 256, 256, 0, stream>>>(W_in, W_out, kpw, WiP, WoP, KpP, stats);

  const float* Xs[2] = { src, tgt };
  const float* Cs[2] = { src_coords, tgt_coords };
  const int*   Nb[2] = { src_nb, tgt_nb };
  const size_t smem_kp = 221440;                     // nf+infl+agg+kpw+kp

  for (int b = 0; b < 2; ++b) {
    k_gemm_in<<<2048, 128, 0, stream>>>(Xs[b], WiP, b_in, Hb[b]);
    k_kpconv<<<N_PTS / 16, 128, smem_kp, stream>>>(Hb[b], Cs[b], Nb[b], kpts, KpP, Gb[b]);
    k_gemm_out<<<8192, 128, 0, stream>>>(Gb[b], WoP, b_out, outs[b], stats + b * 1024);
    k_bn<<<(N_PTS * COUT + 255) / 256, 256, 0, stream>>>(outs[b], stats + b * 1024, gamma, beta);
  }
  k_copy_coords<<<(2 * N_PTS * 3 + 255) / 256, 256, 0, stream>>>(src_coords, tgt_coords, cdst);
}
